// BILSTM_CRF_6038724018944
// MI455X (gfx1250) — compile-verified, matrix operand order checked by reading, not run on hardware
//
#include <hip/hip_runtime.h>
#include <hip/hip_bf16.h>

// ---------------------------------------------------------------------------
// BiLSTM-CRF for MI455X (gfx1250).
//  - Input projections: bf16 WMMA GEMM (v_wmma_f32_16x16x32_bf16, f32 accum)
//    with double-buffered GLOBAL_LOAD_ASYNC_TO_LDS_B128 staging (ASYNCcnt).
//    All 8 fragments of a K-step are preloaded before the 8 WMMAs so ds_load
//    latency is pipelined instead of serialized on recycled VGPRs.
//  - Recurrence: persistent kernel, per-timestep software grid barrier,
//    double-buffered h state, W_hh streamed from L2 each step (float4 GEMV).
//  - Dense + Viterbi: small VALU kernels.
// ---------------------------------------------------------------------------

typedef __attribute__((ext_vector_type(16))) __bf16 v16bf;
typedef __attribute__((ext_vector_type(8)))  __bf16 v8bf;
typedef __attribute__((ext_vector_type(8)))  float  v8f;

#define T_LEN 2048
#define E_DIM 400
#define E_PAD 448              // multiple of BK=64
#define H_DIM 1024
#define G4    4096
#define D2    2048
#define K_TAG 12
#define START_TAG 10
#define STOP_TAG  11
#define NEG_INF  -10000.0f
#define BK    64

// ------------------------------ helpers ------------------------------------

__device__ __forceinline__ __bf16 to_bf16(float f) {
  union { float f; unsigned u; } a; a.f = f;
  unsigned u = a.u;
  unsigned r = u + 0x7FFFu + ((u >> 16) & 1u);   // round-to-nearest-even
  union { unsigned short s; __bf16 b; } o;
  o.s = (unsigned short)(r >> 16);
  return o.b;
}

// CDNA5 async DMA: global memory -> LDS, 16B per lane, tracked by ASYNCcnt.
__device__ __forceinline__ void async_b128(unsigned lds_addr, const void* gaddr) {
  asm volatile("global_load_async_to_lds_b128 %0, %1, off"
               :: "v"(lds_addr), "v"((unsigned long long)(uintptr_t)gaddr)
               : "memory");
}
__device__ __forceinline__ void wait_async0() {
  asm volatile("s_wait_asynccnt 0x0" ::: "memory");
}

// Build one 16x32 bf16 fragment from an LDS tile per the CDNA5 WMMA layout:
// lane L: row = L%16 ; K-chunks {kb..kb+7} and {16+kb..16+kb+7}, kb=(L/16)*8.
__device__ __forceinline__ v16bf load_frag(const __bf16* base, int ld, int lane) {
  const int r  = lane & 15;
  const int kb = (lane >> 4) << 3;
  const __bf16* p = base + r * ld;
  union { v16bf v; v8bf h[2]; } u;
  u.h[0] = *(const v8bf*)(p + kb);
  u.h[1] = *(const v8bf*)(p + 16 + kb);
  return u.v;
}

// --------------------------- conversion kernels ----------------------------

__global__ __launch_bounds__(256)
void cvt_bf16_pad(const float* __restrict__ src, __bf16* __restrict__ dst,
                  int rows, int cols, int colsP) {
  int i = blockIdx.x * 256 + threadIdx.x;
  if (i >= rows * colsP) return;
  int r = i / colsP, c = i - r * colsP;
  float v = (c < cols) ? src[(size_t)r * cols + c] : 0.0f;
  dst[i] = to_bf16(v);
}

__global__ __launch_bounds__(256)
void zero_f32(float* __restrict__ p, int n) {
  int i = blockIdx.x * 256 + threadIdx.x;
  if (i < n) p[i] = 0.0f;
}

// ------------------------------ WMMA GEMM ----------------------------------
// C[M,N] = A[M,Kp] * B[N,Kp]^T + bias0[N] + bias1[N]
// M % 64 == 0, N % 128 == 0, Kp % 64 == 0. Block = 256 threads (8 waves).
// Block tile 64x128, wave tile 32x32, BK=64 => 8 v_wmma per wave / iteration.
// Tiles staged with async-to-LDS DMA, double buffered: one barrier per step.

__global__ __launch_bounds__(256)
void gemm_bf16_wmma(const __bf16* __restrict__ A, const __bf16* __restrict__ B,
                    const float* __restrict__ bias0, const float* __restrict__ bias1,
                    float* __restrict__ C, int M, int N, int Kp) {
  __shared__ __align__(16) __bf16 Asub[2][64][BK];
  __shared__ __align__(16) __bf16 Bsub[2][128][BK];

  const int tid  = threadIdx.x;
  const int lane = tid & 31;
  const int wave = tid >> 5;       // 0..7 (wave32)
  const int wm   = wave & 1;       // 2 waves along M
  const int wn   = wave >> 1;      // 4 waves along N
  const int m0   = blockIdx.y * 64;
  const int n0   = blockIdx.x * 128;

  // Per-thread async staging coordinates.
  const int ar = tid >> 2;             // A row 0..63
  const int ac = (tid & 3) << 4;       // A col {0,16,32,48}   (16 elems = 32B)
  const int br = tid >> 1;             // B row 0..127
  const int bc = (tid & 1) << 5;       // B col {0,32}         (32 elems = 64B)

  auto issue_tile = [&](int kk, int buf) {
    const __bf16* ga = &A[(size_t)(m0 + ar) * Kp + kk + ac];
    unsigned la = (unsigned)(uintptr_t)&Asub[buf][ar][ac];
    async_b128(la,      ga);
    async_b128(la + 16, ga + 8);
    const __bf16* gb = &B[(size_t)(n0 + br) * Kp + kk + bc];
    unsigned lb = (unsigned)(uintptr_t)&Bsub[buf][br][bc];
    async_b128(lb,      gb);
    async_b128(lb + 16, gb + 8);
    async_b128(lb + 32, gb + 16);
    async_b128(lb + 48, gb + 24);
  };

  v8f acc[2][2] = {};

  issue_tile(0, 0);
  const int nIter = Kp / BK;
  for (int it = 0; it < nIter; ++it) {
    const int cur = it & 1;
    wait_async0();          // own async ops for buffer `cur` complete
    __syncthreads();        // -> whole tile resident in LDS for all waves
    if (it + 1 < nIter) issue_tile((it + 1) * BK, 1 - cur);  // overlap w/ math

    // Preload ALL fragments of this K-step (4 A + 4 B, distinct registers) so
    // the 16 ds_load_b128 pipeline ahead of the 8 dependent v_wmma.
    v16bf af[2][2], bfr[2][2];
#pragma unroll
    for (int kc = 0; kc < 2; ++kc) {
#pragma unroll
      for (int i = 0; i < 2; ++i)
        af[kc][i] = load_frag(&Asub[cur][wm * 32 + i * 16][kc * 32], BK, lane);
#pragma unroll
      for (int j = 0; j < 2; ++j)
        bfr[kc][j] = load_frag(&Bsub[cur][wn * 32 + j * 16][kc * 32], BK, lane);
    }
#pragma unroll
    for (int kc = 0; kc < 2; ++kc)
#pragma unroll
      for (int i = 0; i < 2; ++i)
#pragma unroll
        for (int j = 0; j < 2; ++j)
          acc[i][j] = __builtin_amdgcn_wmma_f32_16x16x32_bf16(
              false, af[kc][i], false, bfr[kc][j], (short)0, acc[i][j], false, false);
  }

  // Epilogue: C/D 16x16 f32 layout — VGPR r: lanes 0-15 => M=r, lanes 16-31 => M=8+r.
  const int col  = lane & 15;
  const int rofs = (lane < 16) ? 0 : 8;
#pragma unroll
  for (int i = 0; i < 2; ++i)
#pragma unroll
    for (int j = 0; j < 2; ++j) {
      int rbase = m0 + wm * 32 + i * 16 + rofs;
      int cbase = n0 + wn * 32 + j * 16 + col;
      float bs = bias0[cbase] + bias1[cbase];
#pragma unroll
      for (int r = 0; r < 8; ++r)
        C[(size_t)(rbase + r) * N + cbase] = acc[i][j][r] + bs;
    }
}

// --------------------------- persistent LSTM scan --------------------------
// 128 blocks x 256 threads. Blocks 0..63 = forward, 64..127 = backward.
// Each block owns 16 H-channels (all 4 gates). h is double-buffered in global
// memory so one grid barrier per timestep suffices.

__global__ __launch_bounds__(256)
void lstm_scan(const float* __restrict__ preF, const float* __restrict__ preB,
               const float* __restrict__ whhF, const float* __restrict__ whhB,
               float* __restrict__ out,          // [T, 2H]; fwd -> cols 0..H-1, bwd -> H..2H-1
               float* __restrict__ state,        // per dir: h0[H], h1[H], c[H]
               int* __restrict__ sync, int Tn) {
  const int dir = blockIdx.x >> 6;
  const int b   = blockIdx.x & 63;
  const float* pre = dir ? preB : preF;
  const float* whh = dir ? whhB : whhF;
  float* hbufs   = state + dir * (3 * H_DIM);
  float* c_state = hbufs + 2 * H_DIM;
  int* cnt = sync + dir * 8;
  int* gen = cnt + 4;

  __shared__ __align__(16) float h_lds[H_DIM];
  __shared__ float red[16][4][4];
  __shared__ float gates[16][4];

  const int tid  = threadIdx.x;
  const int chl  = tid >> 4;         // 0..15  local channel
  const int gate = (tid >> 2) & 3;   // 0..3
  const int q    = tid & 3;          // 0..3   k-quarter
  const int ch   = b * 16 + chl;
  const float* wrow = whh + (size_t)(gate * H_DIM + ch) * H_DIM + q * 256;
  const float4* w4 = (const float4*)wrow;

  for (int step = 0; step < Tn; ++step) {
    const int t = dir ? (Tn - 1 - step) : step;
    const float* hr = hbufs + ((step & 1) * H_DIM);        // h from step-1
    float*       hw = hbufs + (((step + 1) & 1) * H_DIM);  // h for this step

    if (step == 0) ((float4*)h_lds)[tid] = make_float4(0.f, 0.f, 0.f, 0.f);
    else           ((float4*)h_lds)[tid] = ((const float4*)hr)[tid];
    // Prefetch next step's pre-activation row segment for this block.
    if (tid < 64 && step + 1 < Tn) {
      const int tn = dir ? (Tn - 2 - step) : (step + 1);
      __builtin_prefetch(&pre[(size_t)tn * G4 + (tid & 3) * H_DIM + b * 16 + (tid >> 2)], 0, 1);
    }
    __syncthreads();

    float s = 0.0f;
    const float4* h4 = (const float4*)(h_lds + q * 256);
#pragma unroll 8
    for (int k = 0; k < 64; ++k) {
      float4 a = w4[k], h = h4[k];
      s = fmaf(a.x, h.x, fmaf(a.y, h.y, fmaf(a.z, h.z, fmaf(a.w, h.w, s))));
    }
    red[chl][gate][q] = s;
    __syncthreads();

    if (tid < 64) {
      const int c2 = tid >> 2, g2 = tid & 3;
      gates[c2][g2] = red[c2][g2][0] + red[c2][g2][1] + red[c2][g2][2] + red[c2][g2][3]
                    + pre[(size_t)t * G4 + g2 * H_DIM + b * 16 + c2];
    }
    __syncthreads();

    if (tid < 16) {
      const float zi = gates[tid][0], zf = gates[tid][1];
      const float zg = gates[tid][2], zo = gates[tid][3];
      const float ig = 1.0f / (1.0f + __expf(-zi));
      const float fg = 1.0f / (1.0f + __expf(-zf));
      const float gg = tanhf(zg);
      const float og = 1.0f / (1.0f + __expf(-zo));
      const int c0 = b * 16 + tid;
      const float cp = (step == 0) ? 0.0f : c_state[c0];
      const float cn = fg * cp + ig * gg;
      const float hn = og * tanhf(cn);
      c_state[c0] = cn;
      hw[c0] = hn;
      out[(size_t)t * D2 + dir * H_DIM + c0] = hn;
    }

    // grid barrier over this direction's 64 blocks
    __threadfence();
    __syncthreads();
    if (tid == 0) {
      int a = atomicAdd(cnt, 1);
      if (a == 63) {
        atomicExch(cnt, 0);
        __threadfence();
        atomicAdd(gen, 1);
      } else {
        while (__hip_atomic_load(gen, __ATOMIC_ACQUIRE, __HIP_MEMORY_SCOPE_AGENT) <= step)
          __builtin_amdgcn_s_sleep(2);
      }
    }
    __syncthreads();
  }
}

// ------------------------------ dense [T,K] --------------------------------

__global__ __launch_bounds__(256)
void dense12(const float* __restrict__ hcat, const float* __restrict__ w,
             const float* __restrict__ bias, float* __restrict__ feats) {
  int i = blockIdx.x * 256 + threadIdx.x;     // t*K + j
  if (i >= T_LEN * K_TAG) return;
  int t = i / K_TAG, j = i - t * K_TAG;
  const float4* hrow = (const float4*)(hcat + (size_t)t * D2);
  const float4* wrow = (const float4*)(w + (size_t)j * D2);
  float s = bias[j];
  for (int k = 0; k < D2 / 4; ++k) {
    float4 a = hrow[k], bv = wrow[k];
    s = fmaf(a.x, bv.x, fmaf(a.y, bv.y, fmaf(a.z, bv.z, fmaf(a.w, bv.w, s))));
  }
  feats[i] = s;
}

// ------------------------------ Viterbi ------------------------------------

__global__ __launch_bounds__(32)
void viterbi(const float* __restrict__ feats, const float* __restrict__ trans,
             int* __restrict__ bp, float* __restrict__ out) {
  __shared__ float fv[K_TAG], nfv[K_TAG], trs[K_TAG * K_TAG];
  const int tid = threadIdx.x;
  for (int i = tid; i < K_TAG * K_TAG; i += 32) trs[i] = trans[i];
  if (tid < K_TAG) fv[tid] = (tid == START_TAG) ? 0.0f : NEG_INF;
  __syncthreads();

  for (int t = 0; t < T_LEN; ++t) {
    if (tid < K_TAG) {
      float best = fv[0] + trs[tid * K_TAG];
      int arg = 0;
#pragma unroll
      for (int j = 1; j < K_TAG; ++j) {
        float v = fv[j] + trs[tid * K_TAG + j];
        if (v > best) { best = v; arg = j; }
      }
      bp[t * K_TAG + tid] = arg;
      nfv[tid] = best + feats[t * K_TAG + tid];
    }
    __syncthreads();
    if (tid < K_TAG) fv[tid] = nfv[tid];
    __syncthreads();
  }

  if (tid == 0) {
    float best = fv[0] + trs[STOP_TAG * K_TAG];
    int arg = 0;
    for (int i = 1; i < K_TAG; ++i) {
      float v = fv[i] + trs[STOP_TAG * K_TAG + i];
      if (v > best) { best = v; arg = i; }
    }
    out[0] = best;
    int tag = arg;
    for (int t = T_LEN - 1; t >= 0; --t) {
      out[1 + t] = (float)tag;
      tag = bp[t * K_TAG + tag];
    }
  }
}

// ------------------------------ launcher -----------------------------------

extern "C" void kernel_launch(void* const* d_in, const int* in_sizes, int n_in,
                              void* d_out, int out_size, void* d_ws, size_t ws_size,
                              hipStream_t stream) {
  (void)in_sizes; (void)n_in; (void)out_size; (void)ws_size;
  const float* x     = (const float*)d_in[0];
  const float* wif0  = (const float*)d_in[1];
  const float* whf0  = (const float*)d_in[2];
  const float* bif0  = (const float*)d_in[3];
  const float* bhf0  = (const float*)d_in[4];
  const float* wib0  = (const float*)d_in[5];
  const float* whb0  = (const float*)d_in[6];
  const float* bib0  = (const float*)d_in[7];
  const float* bhb0  = (const float*)d_in[8];
  const float* wif1  = (const float*)d_in[9];
  const float* whf1  = (const float*)d_in[10];
  const float* bif1  = (const float*)d_in[11];
  const float* bhf1  = (const float*)d_in[12];
  const float* wib1  = (const float*)d_in[13];
  const float* whb1  = (const float*)d_in[14];
  const float* bib1  = (const float*)d_in[15];
  const float* bhb1  = (const float*)d_in[16];
  const float* dw    = (const float*)d_in[17];
  const float* db    = (const float*)d_in[18];
  const float* trans = (const float*)d_in[19];
  float* out = (float*)d_out;

  char* w = (char*)d_ws;
  size_t o = 0;
  auto take = [&](size_t bytes) -> char* {
    char* p = w + o;
    o = (o + bytes + 255) & ~(size_t)255;
    return p;
  };
  __bf16* XBF0 = (__bf16*)take((size_t)T_LEN * E_PAD * 2);
  __bf16* WIF0 = (__bf16*)take((size_t)G4 * E_PAD * 2);
  __bf16* WIB0 = (__bf16*)take((size_t)G4 * E_PAD * 2);
  __bf16* XBF1 = (__bf16*)take((size_t)T_LEN * D2 * 2);
  __bf16* WIF1 = (__bf16*)take((size_t)G4 * D2 * 2);
  __bf16* WIB1 = (__bf16*)take((size_t)G4 * D2 * 2);
  float*  PREF = (float*)take((size_t)T_LEN * G4 * 4);
  float*  PREB = (float*)take((size_t)T_LEN * G4 * 4);
  float*  X1   = (float*)take((size_t)T_LEN * D2 * 4);
  float*  HCAT = (float*)take((size_t)T_LEN * D2 * 4);
  float*  FEAT = (float*)take((size_t)T_LEN * K_TAG * 4);
  float*  STAT = (float*)take((size_t)2 * 3 * H_DIM * 4);
  int*    SYNC = (int*)take(64 * 4);
  int*    BP   = (int*)take((size_t)T_LEN * K_TAG * 4);

  const dim3 blk(256);
  auto cdiv = [](int a, int b) { return (a + b - 1) / b; };

  // ---- layer 0 input projections (WMMA) ----
  cvt_bf16_pad<<<cdiv(T_LEN * E_PAD, 256), blk, 0, stream>>>(x, XBF0, T_LEN, E_DIM, E_PAD);
  cvt_bf16_pad<<<cdiv(G4 * E_PAD, 256), blk, 0, stream>>>(wif0, WIF0, G4, E_DIM, E_PAD);
  cvt_bf16_pad<<<cdiv(G4 * E_PAD, 256), blk, 0, stream>>>(wib0, WIB0, G4, E_DIM, E_PAD);
  gemm_bf16_wmma<<<dim3(G4 / 128, T_LEN / 64), blk, 0, stream>>>(
      XBF0, WIF0, bif0, bhf0, PREF, T_LEN, G4, E_PAD);
  gemm_bf16_wmma<<<dim3(G4 / 128, T_LEN / 64), blk, 0, stream>>>(
      XBF0, WIB0, bib0, bhb0, PREB, T_LEN, G4, E_PAD);

  // ---- layer 0 recurrence ----
  zero_f32<<<1, 64, 0, stream>>>((float*)SYNC, 64);
  lstm_scan<<<128, blk, 0, stream>>>(PREF, PREB, whf0, whb0, X1, STAT, SYNC, T_LEN);

  // ---- layer 1 input projections (WMMA) ----
  cvt_bf16_pad<<<cdiv(T_LEN * D2, 256), blk, 0, stream>>>(X1, XBF1, T_LEN, D2, D2);
  cvt_bf16_pad<<<cdiv(G4 * D2, 256), blk, 0, stream>>>(wif1, WIF1, G4, D2, D2);
  cvt_bf16_pad<<<cdiv(G4 * D2, 256), blk, 0, stream>>>(wib1, WIB1, G4, D2, D2);
  gemm_bf16_wmma<<<dim3(G4 / 128, T_LEN / 64), blk, 0, stream>>>(
      XBF1, WIF1, bif1, bhf1, PREF, T_LEN, G4, D2);
  gemm_bf16_wmma<<<dim3(G4 / 128, T_LEN / 64), blk, 0, stream>>>(
      XBF1, WIB1, bib1, bhb1, PREB, T_LEN, G4, D2);

  // ---- layer 1 recurrence ----
  zero_f32<<<1, 64, 0, stream>>>((float*)SYNC, 64);
  lstm_scan<<<128, blk, 0, stream>>>(PREF, PREB, whf1, whb1, HCAT, STAT, SYNC, T_LEN);

  // ---- dense + Viterbi ----
  dense12<<<cdiv(T_LEN * K_TAG, 256), blk, 0, stream>>>(HCAT, dw, db, FEAT);
  viterbi<<<1, 32, 0, stream>>>(FEAT, trans, BP, out);
}
